// FLASH_43937515438729
// MI455X (gfx1250) — matrix operand order repaired and checked by reading, not compiled, not run on hardware
//
#include <hip/hip_runtime.h>
#include <math.h>

// ---------------------------------------------------------------------------
// FLASH / GAU block for MI455X (gfx1250, wave32, WMMA bf16, async G->LDS)
// ---------------------------------------------------------------------------

#define DIMX  1024
#define HIDX  2048
#define QKX   128
#define GX    256
#define BX    4
#define NX    4096
#define ROWS  (BX * NX)        // 16384
#define NGX   (NX / GX)        // 16
#define NGRP  (BX * NGX)       // 64

typedef __attribute__((ext_vector_type(16))) __bf16 bf16x16;
typedef __attribute__((ext_vector_type(8)))  __bf16 bf16x8;
typedef __attribute__((ext_vector_type(8)))  float  v8f;

__device__ __forceinline__ __bf16 f2bf(float f) {
  unsigned u = __builtin_bit_cast(unsigned, f);
  u += 0x7FFFu + ((u >> 16) & 1u);           // round-to-nearest-even
  unsigned short h = (unsigned short)(u >> 16);
  return __builtin_bit_cast(__bf16, h);
}
__device__ __forceinline__ float bf2f(__bf16 b) {
  unsigned short h = __builtin_bit_cast(unsigned short, b);
  unsigned u = ((unsigned)h) << 16;
  return __builtin_bit_cast(float, u);
}
__device__ __forceinline__ float silu_f(float v) { return v / (1.0f + __expf(-v)); }

// Async global -> LDS 16-byte copy (per-lane), tracked by ASYNCcnt.
// vdst = 32-bit LDS byte offset, vaddr = 64-bit global address (GV mode).
__device__ __forceinline__ void async_g2l_b128(const __bf16* gptr, __bf16* lptr) {
  unsigned lds_off = (unsigned)(unsigned long long)(uintptr_t)lptr;
  asm volatile("global_load_async_to_lds_b128 %0, %1, off"
               :: "v"(lds_off), "v"(gptr) : "memory");
}
__device__ __forceinline__ void wait_async0() {
  asm volatile("s_wait_asynccnt 0x0" ::: "memory");
}

// ---------------------------------------------------------------------------
// Small helper kernels
// ---------------------------------------------------------------------------
__global__ void cvt_f32_bf16(const float* __restrict__ in, __bf16* __restrict__ out, int n) {
  int i = blockIdx.x * blockDim.x + threadIdx.x;
  if (i < n) out[i] = f2bf(in[i]);
}

// Relative position bias depends only on d = i - j in [-255,255] -> 511 table
__global__ void bias_table_kernel(const float* __restrict__ rel_emb, float* __restrict__ table) {
  int t = blockIdx.x * blockDim.x + threadIdx.x;
  if (t >= 511) return;
  int n = t - 255;                         // n = i - j
  int ret = (n < 0) ? 16 : 0;              // NUM_BUCKETS/2 = 16
  int a = (n < 0) ? -n : n;
  int add;
  if (a < 8) {
    add = a;
  } else {
    int vl = 8 + (int)(logf((float)a / 8.0f) / logf(16.0f) * 8.0f);
    add = (vl < 15) ? vl : 15;
  }
  table[t] = rel_emb[ret + add] * 11.313708498984761f;   // * sqrt(QK)
}

// LayerNorm of one 1024-wide row per block; writes bf16
__global__ __launch_bounds__(256) void layernorm_bf16(
    const float* __restrict__ x, const float* __restrict__ g,
    const float* __restrict__ b, __bf16* __restrict__ out) {
  __shared__ float red[256];
  const int r = blockIdx.x;
  const int t = threadIdx.x;
  const float* xr = x + (size_t)r * DIMX;
  float v[4];
  float s = 0.f;
#pragma unroll
  for (int i = 0; i < 4; ++i) { v[i] = xr[t + 256 * i]; s += v[i]; }
  red[t] = s; __syncthreads();
  for (int o = 128; o > 0; o >>= 1) { if (t < o) red[t] += red[t + o]; __syncthreads(); }
  const float mu = red[0] * (1.0f / DIMX);
  __syncthreads();
  float vs = 0.f;
#pragma unroll
  for (int i = 0; i < 4; ++i) { float d = v[i] - mu; vs += d * d; }
  red[t] = vs; __syncthreads();
  for (int o = 128; o > 0; o >>= 1) { if (t < o) red[t] += red[t + o]; __syncthreads(); }
  const float rstd = rsqrtf(red[0] * (1.0f / DIMX) + 1e-5f);
#pragma unroll
  for (int i = 0; i < 4; ++i) {
    int c = t + 256 * i;
    out[(size_t)r * DIMX + c] = f2bf((v[i] - mu) * rstd * g[c] + b[c]);
  }
}

// ---------------------------------------------------------------------------
// Generic bf16 WMMA GEMM: C = A(MxK) * B(KxN), f32 accumulate, fused epilogues
// Block = 256 threads = 8 wave32 waves; tile 128x128, BK=32.
// Wave (wm in 0..3, wn in 0..1) owns a 32x64 sub-tile = 2x4 WMMA accumulators.
// Contiguous staging paths use GLOBAL_LOAD_ASYNC_TO_LDS_B128 (ASYNCcnt);
// transpose-scatter paths stage through VGPRs.
// ---------------------------------------------------------------------------
enum { EPI_HID = 0, EPI_QK = 1, EPI_SIM = 2, EPI_BF16 = 3,
       EPI_SCALEBF = 4, EPI_FUSE = 5, EPI_FINAL = 6 };

struct GemmP {
  const __bf16* A;
  const __bf16* B;
  long long sAz, sBz, sCz;       // per-blockIdx.z element strides
  int M, N, K;
  float scale;
  const float* bias;             // b_h / b_qk / b_out / rel-pos table
  const float* gamma;            // os_gamma (flattened 4x128)
  const float* beta;             // os_beta
  const __bf16* gate;            // EPI_FUSE
  const __bf16* qout;            // EPI_FUSE
  const float* resid;            // EPI_FINAL
  __bf16* obf0; __bf16* obf1; __bf16* obf2; __bf16* obf3;
  float* of32;
};

template <int EPI, bool TA, bool TB>
__global__ __launch_bounds__(256) void wmma_gemm(GemmP p) {
  __shared__ __bf16 As[128][48];   // [m][k], padded: 16B-aligned vector access
  __shared__ __bf16 Bs[128][48];   // [n][k] (B stored K-transposed in LDS)

  const int tid  = threadIdx.x;
  const int wid  = tid >> 5;
  const int lane = tid & 31;
  const int wm   = wid & 3;        // wave M index (x32 rows)
  const int wn   = wid >> 2;       // wave N index (x64 cols)
  const int m0   = blockIdx.x * 128;
  const int n0   = blockIdx.y * 128;
  const int z    = blockIdx.z;

  const __bf16* Ab = p.A + (size_t)z * (size_t)p.sAz;
  const __bf16* Bb = p.B + (size_t)z * (size_t)p.sBz;

  v8f acc[2][4] = {};

  const int laneRow  = lane & 15;
  const int laneHalf = lane >> 4;        // 0 or 1

  for (int k0 = 0; k0 < p.K; k0 += 32) {
    __syncthreads();
    // ---- load A tile -> As[m][k]
    if constexpr (!TA) {
      const int row = tid >> 1, ch = (tid & 1) << 4;
      const __bf16* g = Ab + (size_t)(m0 + row) * p.K + (k0 + ch);
      async_g2l_b128(g,     &As[row][ch]);       // global_load_async_to_lds_b128
      async_g2l_b128(g + 8, &As[row][ch + 8]);
      if (k0 + 32 < p.K) __builtin_prefetch(g + 32, 0, 3);   // global_prefetch_b8
    } else {
      // A supplied as K x M row-major (used for lin_k^T @ v): transpose scatter
      const int kk = tid >> 3, mc = (tid & 7) << 4;
      const __bf16* g = Ab + (size_t)(k0 + kk) * p.M + (m0 + mc);
      bf16x8 x0 = *(const bf16x8*)g;
      bf16x8 x1 = *(const bf16x8*)(g + 8);
#pragma unroll
      for (int i = 0; i < 8; ++i) { As[mc + i][kk] = x0[i]; As[mc + 8 + i][kk] = x1[i]; }
    }
    // ---- load B tile -> Bs[n][k]
    if constexpr (TB) {
      // B supplied as N x K row-major (used for q @ k^T): straight async copy
      const int row = tid >> 1, ch = (tid & 1) << 4;
      const __bf16* g = Bb + (size_t)(n0 + row) * p.K + (k0 + ch);
      async_g2l_b128(g,     &Bs[row][ch]);
      async_g2l_b128(g + 8, &Bs[row][ch + 8]);
    } else {
      const int kk = tid >> 3, nc = (tid & 7) << 4;
      const __bf16* g = Bb + (size_t)(k0 + kk) * p.N + (n0 + nc);
      bf16x8 x0 = *(const bf16x8*)g;
      bf16x8 x1 = *(const bf16x8*)(g + 8);
#pragma unroll
      for (int i = 0; i < 8; ++i) { Bs[nc + i][kk] = x0[i]; Bs[nc + 8 + i][kk] = x1[i]; }
      if (k0 + 32 < p.K) __builtin_prefetch(g + (size_t)32 * p.N, 0, 3);
    }
    // Drain ASYNCcnt before signalling the barrier so consumers see the tile.
    if constexpr (!TA || TB) wait_async0();
    __syncthreads();

    // ---- A fragments (16-bit A layout: lanes 0-15 K{0-7,16-23}, lanes 16-31 K{8-15,24-31})
    bf16x16 af[2];
#pragma unroll
    for (int mi = 0; mi < 2; ++mi) {
      const int r  = wm * 32 + mi * 16 + laneRow;
      const int ko = laneHalf * 8;
      bf16x8 lo = *(const bf16x8*)&As[r][ko];
      bf16x8 hi = *(const bf16x8*)&As[r][16 + ko];
      af[mi] = __builtin_shufflevector(lo, hi, 0,1,2,3,4,5,6,7,8,9,10,11,12,13,14,15);
    }
    // ---- B fragments (per-lane column, same K order as A: K{0-7,16-23} / K{8-15,24-31}) and WMMA
#pragma unroll
    for (int ni = 0; ni < 4; ++ni) {
      const int c  = wn * 64 + ni * 16 + laneRow;
      const int kb = laneHalf * 8;
      bf16x8 lo = *(const bf16x8*)&Bs[c][kb];
      bf16x8 hi = *(const bf16x8*)&Bs[c][16 + kb];
      bf16x16 bfrag = __builtin_shufflevector(lo, hi, 0,1,2,3,4,5,6,7,8,9,10,11,12,13,14,15);
#pragma unroll
      for (int mi = 0; mi < 2; ++mi) {
        acc[mi][ni] = __builtin_amdgcn_wmma_f32_16x16x32_bf16(
            false, af[mi], false, bfrag, (short)0, acc[mi][ni], false, false);
      }
    }
  }

  // ---- epilogue: C/D layout -> lane holds col (lane&15), rows (lane<16?0:8)+j
#pragma unroll
  for (int mi = 0; mi < 2; ++mi) {
#pragma unroll
    for (int ni = 0; ni < 4; ++ni) {
      const int rbase = m0 + wm * 32 + mi * 16 + laneHalf * 8;
      const int c     = n0 + wn * 64 + ni * 16 + laneRow;
#pragma unroll
      for (int j = 0; j < 8; ++j) {
        const int r = rbase + j;
        const float val = acc[mi][ni][j];
        const size_t idx = (size_t)z * (size_t)p.sCz + (size_t)r * p.N + c;
        if constexpr (EPI == EPI_HID) {
          float h = silu_f(val + p.bias[c]);
          if (c < HIDX) p.obf0[(size_t)r * HIDX + c]          = f2bf(h);  // v
          else          p.obf1[(size_t)r * HIDX + (c - HIDX)] = f2bf(h);  // gate
        } else if constexpr (EPI == EPI_QK) {
          float h = silu_f(val + p.bias[c]);
          const size_t o = (size_t)r * QKX + c;
          p.obf0[o] = f2bf(h * p.gamma[c]            + p.beta[c]);             // quad_q
          p.obf1[o] = f2bf(h * p.gamma[QKX + c]      + p.beta[QKX + c]);       // lin_q
          p.obf2[o] = f2bf(h * p.gamma[2 * QKX + c]  + p.beta[2 * QKX + c]);   // quad_k
          p.obf3[o] = f2bf(h * p.gamma[3 * QKX + c]  + p.beta[3 * QKX + c]);   // lin_k
        } else if constexpr (EPI == EPI_SIM) {
          float s = val * p.scale + p.bias[r - c + 255];   // r,c local to group
          s = s > 0.f ? s : 0.f;
          p.obf0[idx] = f2bf(s * s);                        // relu(sim)^2
        } else if constexpr (EPI == EPI_BF16) {
          p.obf0[idx] = f2bf(val);
        } else if constexpr (EPI == EPI_SCALEBF) {
          p.obf0[idx] = f2bf(val * p.scale);
        } else if constexpr (EPI == EPI_FUSE) {
          const float q = bf2f(p.qout[idx]);
          const float g = bf2f(p.gate[idx]);
          p.obf0[idx] = f2bf(g * (q + val));
        } else { // EPI_FINAL
          p.of32[idx] = val + p.bias[c] + p.resid[idx];
        }
      }
    }
  }
}

// ---------------------------------------------------------------------------
// Host side
// ---------------------------------------------------------------------------
extern "C" void kernel_launch(void* const* d_in, const int* in_sizes, int n_in,
                              void* d_out, int out_size, void* d_ws, size_t ws_size,
                              hipStream_t stream) {
  (void)in_sizes; (void)n_in; (void)out_size; (void)ws_size;
  const float* x       = (const float*)d_in[0];
  const float* ln_g    = (const float*)d_in[1];
  const float* ln_b    = (const float*)d_in[2];
  const float* W_h     = (const float*)d_in[3];
  const float* b_h     = (const float*)d_in[4];
  const float* W_qk    = (const float*)d_in[5];
  const float* b_qk    = (const float*)d_in[6];
  const float* os_g    = (const float*)d_in[7];
  const float* os_b    = (const float*)d_in[8];
  const float* rel_emb = (const float*)d_in[9];
  const float* W_out   = (const float*)d_in[10];
  const float* b_out   = (const float*)d_in[11];

  char* ws = (char*)d_ws;
  size_t off = 0;
  auto alloc = [&](size_t bytes) -> void* {
    off = (off + 255) & ~(size_t)255;
    void* p = ws + off;
    off += bytes;
    return p;
  };

  __bf16* normed = (__bf16*)alloc((size_t)ROWS * DIMX * 2);
  __bf16* Whb    = (__bf16*)alloc((size_t)DIMX * 2 * HIDX * 2);
  __bf16* Wqkb   = (__bf16*)alloc((size_t)DIMX * QKX * 2);
  __bf16* Woutb  = (__bf16*)alloc((size_t)HIDX * DIMX * 2);
  __bf16* vbuf   = (__bf16*)alloc((size_t)ROWS * HIDX * 2);
  __bf16* gate   = (__bf16*)alloc((size_t)ROWS * HIDX * 2);
  __bf16* qq     = (__bf16*)alloc((size_t)ROWS * QKX * 2);   // quad_q
  __bf16* lq     = (__bf16*)alloc((size_t)ROWS * QKX * 2);   // lin_q
  __bf16* qk     = (__bf16*)alloc((size_t)ROWS * QKX * 2);   // quad_k
  __bf16* lk     = (__bf16*)alloc((size_t)ROWS * QKX * 2);   // lin_k
  __bf16* attn   = (__bf16*)alloc((size_t)NGRP * GX * GX * 2);
  __bf16* qout   = (__bf16*)alloc((size_t)ROWS * HIDX * 2);
  __bf16* linkv  = (__bf16*)alloc((size_t)BX * QKX * HIDX * 2);
  __bf16* comb   = (__bf16*)alloc((size_t)ROWS * HIDX * 2);
  float*  btab   = (float*)alloc(511 * 4);

  // --- weight conversion + bias table + layernorm
  cvt_f32_bf16<<<(DIMX * 2 * HIDX + 255) / 256, 256, 0, stream>>>(W_h, Whb, DIMX * 2 * HIDX);
  cvt_f32_bf16<<<(DIMX * QKX + 255) / 256, 256, 0, stream>>>(W_qk, Wqkb, DIMX * QKX);
  cvt_f32_bf16<<<(HIDX * DIMX + 255) / 256, 256, 0, stream>>>(W_out, Woutb, HIDX * DIMX);
  bias_table_kernel<<<2, 256, 0, stream>>>(rel_emb, btab);
  layernorm_bf16<<<ROWS, 256, 0, stream>>>(x, ln_g, ln_b, normed);

  // --- GEMM1: normed @ W_h -> silu -> v, gate
  {
    GemmP p{}; p.A = normed; p.B = Whb;
    p.M = ROWS; p.N = 2 * HIDX; p.K = DIMX;
    p.bias = b_h; p.obf0 = vbuf; p.obf1 = gate;
    wmma_gemm<EPI_HID, false, false><<<dim3(ROWS / 128, (2 * HIDX) / 128, 1), 256, 0, stream>>>(p);
  }
  // --- GEMM qk: normed @ W_qk -> silu -> 4 heads via gamma/beta
  {
    GemmP p{}; p.A = normed; p.B = Wqkb;
    p.M = ROWS; p.N = QKX; p.K = DIMX;
    p.bias = b_qk; p.gamma = os_g; p.beta = os_b;
    p.obf0 = qq; p.obf1 = lq; p.obf2 = qk; p.obf3 = lk;
    wmma_gemm<EPI_QK, false, false><<<dim3(ROWS / 128, 1, 1), 256, 0, stream>>>(p);
  }
  // --- sim per (b,g): quad_q @ quad_k^T / G + bias, relu^2 -> attn
  {
    GemmP p{}; p.A = qq; p.B = qk;                 // B is N x K (k^T free via TB path)
    p.M = GX; p.N = GX; p.K = QKX;
    p.sAz = GX * QKX; p.sBz = GX * QKX; p.sCz = GX * GX;
    p.scale = 1.0f / GX; p.bias = btab; p.obf0 = attn;
    wmma_gemm<EPI_SIM, false, true><<<dim3(GX / 128, GX / 128, NGRP), 256, 0, stream>>>(p);
  }
  // --- quad_out per (b,g): attn @ v_g
  {
    GemmP p{}; p.A = attn; p.B = vbuf;
    p.M = GX; p.N = HIDX; p.K = GX;
    p.sAz = GX * GX; p.sBz = (long long)GX * HIDX; p.sCz = (long long)GX * HIDX;
    p.obf0 = qout;
    wmma_gemm<EPI_BF16, false, false><<<dim3(GX / 128, HIDX / 128, NGRP), 256, 0, stream>>>(p);
  }
  // --- lin_kv per batch: lin_k^T @ v / N   (A transposed path)
  {
    GemmP p{}; p.A = lk; p.B = vbuf;
    p.M = QKX; p.N = HIDX; p.K = NX;
    p.sAz = (long long)NX * QKX; p.sBz = (long long)NX * HIDX; p.sCz = (long long)QKX * HIDX;
    p.scale = 1.0f / NX; p.obf0 = linkv;
    wmma_gemm<EPI_SCALEBF, true, false><<<dim3(QKX / 128, HIDX / 128, BX), 256, 0, stream>>>(p);
  }
  // --- lin_out per batch: lin_q @ lin_kv ; fused comb = gate*(quad_out + lin_out)
  {
    GemmP p{}; p.A = lq; p.B = linkv;
    p.M = NX; p.N = HIDX; p.K = QKX;
    p.sAz = (long long)NX * QKX; p.sBz = (long long)QKX * HIDX; p.sCz = (long long)NX * HIDX;
    p.gate = gate; p.qout = qout; p.obf0 = comb;
    wmma_gemm<EPI_FUSE, false, false><<<dim3(NX / 128, HIDX / 128, BX), 256, 0, stream>>>(p);
  }
  // --- final: comb @ W_out + b_out + x
  {
    GemmP p{}; p.A = comb; p.B = Woutb;
    p.M = ROWS; p.N = DIMX; p.K = HIDX;
    p.bias = b_out; p.resid = x; p.of32 = (float*)d_out;
    wmma_gemm<EPI_FINAL, false, false><<<dim3(ROWS / 128, DIMX / 128, 1), 256, 0, stream>>>(p);
  }
}
